// GRUModel_38508676776098
// MI455X (gfx1250) — matrix operand order repaired and checked by reading, not compiled, not run on hardware
//
#include <hip/hip_runtime.h>
#include <hip/hip_bf16.h>

// ---------------------------------------------------------------------------
// CDNA5 / gfx1250 implementation. wave32, WMMA f32_16x16x32_f16.
//
// Input flattening assumption (setup_inputs dict order, params in insertion
// order):
//   0: src [8,128,512] f32
//   1: hour_emb [50,512]   2: day_emb [8,512]
//   3+14*layer: wq_w, wq_b, wk_w, wk_b, wv_w, wv_b, ffa_w, ffa_b, ffb_w,
//               ffb_b, ln1_g, ln1_b, ln2_g, ln2_b        (layer = 0,1)
//   31: dec_poi_w [1024,5000]  32: dec_poi_b  33: dec_cat_w [1024,300]  34: dec_cat_b
//   35: batch_seq_lens [8] i32   36: ts [8,128] i32   37: label_ts [8,128] i32
// Output: out_poi [8,128,5000] f32 then out_cat [8,128,300] f32, flat concat.
//
// All GEMM N-dims are padded to multiples of 128 (zero-filled transposed
// weights) so the WMMA inner loops are completely branchless and EXEC is
// all-ones at every v_wmma.
// ---------------------------------------------------------------------------

#define Bb   8
#define Ll   128
#define Hh   512
#define Mm   (Bb * Ll)          // 1024
#define TUc  86400
#define NEGV (-4294967295.0f)

typedef __attribute__((ext_vector_type(16))) _Float16 v16h;
typedef __attribute__((ext_vector_type(8)))  _Float16 v8h;
typedef __attribute__((ext_vector_type(8)))  float    v8f;

union Frag16 { v16h v; v8h h[2]; };

// A-fragment (16x32 f16): lane holds row m = lane&15; two contiguous 8-half
// runs at k = koff and 16+koff, koff = (lane<16)?0:8.
__device__ __forceinline__ v16h frag_a(const _Float16* rowp, int koff) {
    Frag16 f;
    f.h[0] = *(const v8h*)(rowp + koff);
    f.h[1] = *(const v8h*)(rowp + 16 + koff);
    return f.v;
}
// B-fragment (32x16 f16): lane holds col n = lane&15; two contiguous 8-half
// runs at k = kb and 16+kb, kb = (lane<16)?0:8.  Source must be [N][K] (K-contiguous).
__device__ __forceinline__ v16h frag_b(const _Float16* rowp, int kb) {
    Frag16 f;
    f.h[0] = *(const v8h*)(rowp + kb);
    f.h[1] = *(const v8h*)(rowp + kb + 16);
    return f.v;
}

__device__ __forceinline__ v8f wmma_f16(v16h a, v16h b, v8f c) {
    return __builtin_amdgcn_wmma_f32_16x16x32_f16(false, a, false, b,
                                                  (short)0, c, false, false);
}

// ---------------------------------------------------------------------------
// GEMM:  C[M,N] = A16[M,K] @ Bt16[Npad,K]^T (+bias)(+relu),  Npad % 128 == 0.
// Block = 256 threads (8 waves).  Block tile 64(M) x 128(N); wave tile 16x64.
// K % 64 == 0; inner loop is 2x software-pipelined, branchless.
// ---------------------------------------------------------------------------
__global__ __launch_bounds__(256)
void k_gemm(const _Float16* __restrict__ A, const _Float16* __restrict__ Bt,
            const float* __restrict__ bias,
            float* __restrict__ Cf, _Float16* __restrict__ C16,
            _Float16* __restrict__ C16t,
            int N, int K, int relu)
{
    const int lane  = (int)threadIdx.x & 31;
    const int wave  = __builtin_amdgcn_readfirstlane((int)threadIdx.x) >> 5;
    const int waveM = wave & 3;
    const int waveN = wave >> 2;
    const int m0 = blockIdx.y * 64 + waveM * 16;
    const int n0 = blockIdx.x * 128 + waveN * 64;
    const int n15  = lane & 15;
    const int koff = (lane < 16) ? 0 : 8;
    const int kb   = (lane < 16) ? 0 : 8;

    v8f acc[4] = {};
    const _Float16* arow = A + (size_t)(m0 + n15) * K;
    const _Float16* brow[4];
#pragma unroll
    for (int t = 0; t < 4; ++t)
        brow[t] = Bt + (size_t)(n0 + 16 * t + n15) * K;

    for (int kc = 0; kc < K; kc += 64) {
        v16h fa0 = frag_a(arow + kc, koff);
        v16h fa1 = frag_a(arow + kc + 32, koff);
        v16h fb0[4], fb1[4];
#pragma unroll
        for (int t = 0; t < 4; ++t) fb0[t] = frag_b(brow[t] + kc, kb);
#pragma unroll
        for (int t = 0; t < 4; ++t) fb1[t] = frag_b(brow[t] + kc + 32, kb);
#pragma unroll
        for (int t = 0; t < 4; ++t) acc[t] = wmma_f16(fa0, fb0[t], acc[t]);
#pragma unroll
        for (int t = 0; t < 4; ++t) acc[t] = wmma_f16(fa1, fb1[t], acc[t]);
    }

    const int mhi = (lane < 16) ? 0 : 8;
#pragma unroll
    for (int t = 0; t < 4; ++t) {
        int ncol = n0 + 16 * t + n15;
        float bv = bias ? ((ncol < N) ? bias[ncol] : 0.0f) : 0.0f;
#pragma unroll
        for (int r = 0; r < 8; ++r) {
            int m = m0 + mhi + r;
            float v = acc[t][r] + bv;
            if (relu) v = fmaxf(v, 0.0f);
            if (Cf  && ncol < N) Cf [(size_t)m * N + ncol] = v;
            if (C16 && ncol < N) C16[(size_t)m * N + ncol] = (_Float16)v;
            if (C16t && ncol < N) {                          // V^T: [B][H][L]
                int bb = m >> 7, jj = m & 127;
                C16t[((size_t)bb * Hh + ncol) * Ll + jj] = (_Float16)v;
            }
        }
    }
}

// ---------------------------------------------------------------------------
// Attention: one block per (batch, 16-query tile).  8 waves.
// Phase 1: S = Q K^T via WMMA (each wave owns a 16-key tile), + gathered
//          interval terms from QT (= Q @ [hour;day]^T), + mask -> LDS.
// Phase 2: row softmax in LDS, probabilities -> f16 LDS.
// Phase 3: O = P @ V via WMMA (A-fragments from LDS, B from V^T global).
// ---------------------------------------------------------------------------
__global__ __launch_bounds__(256)
void k_attn(const _Float16* __restrict__ Q16, const _Float16* __restrict__ K16,
            const _Float16* __restrict__ Vt16,
            const float* __restrict__ QT,          // [M][128]: hour 0..63, day 64..127
            const int* __restrict__ ts, const int* __restrict__ lens,
            float* __restrict__ O)
{
    __shared__ __align__(16) float    S [16][Ll];
    __shared__ __align__(16) _Float16 SH[16][Ll];
    __shared__ float redm[16][16];
    __shared__ float reds[16][16];

    const int b    = blockIdx.y;
    const int j0   = blockIdx.x * 16;
    const int lane = (int)threadIdx.x & 31;
    const int wave = __builtin_amdgcn_readfirstlane((int)threadIdx.x) >> 5;
    const int n15  = lane & 15;
    const int koff = (lane < 16) ? 0 : 8;
    const int kb   = (lane < 16) ? 0 : 8;
    const int mhi  = (lane < 16) ? 0 : 8;
    const int len  = lens[b];

    // ---- Phase 1: scores -------------------------------------------------
    {
        const int k0 = wave * 16;
        v8f acc = {};
        const _Float16* qrow = Q16 + (size_t)(b * Ll + j0 + n15) * Hh;
        const _Float16* krow = K16 + (size_t)(b * Ll + k0 + n15) * Hh;
        for (int kc = 0; kc < Hh; kc += 64) {
            v16h fa0 = frag_a(qrow + kc, koff);
            v16h fb0 = frag_b(krow + kc, kb);
            v16h fa1 = frag_a(qrow + kc + 32, koff);
            v16h fb1 = frag_b(krow + kc + 32, kb);
            acc = wmma_f16(fa0, fb0, acc);
            acc = wmma_f16(fa1, fb1, acc);
        }
        const int key  = k0 + n15;
        const int tkey = ts[b * Ll + key];
#pragma unroll
        for (int r = 0; r < 8; ++r) {
            int jl = mhi + r;
            int j  = j0 + jl;
            int d  = ts[b * Ll + j] - tkey;
            bool valid = (j >= key) && (j < len);
            int ih = 0, id = 0;
            if (valid) {
                ih = (b == j) ? 1 : (d % TUc) / 1800 + 2;   // faithful quirk
                id = d / TUc + 1; if (id > 7) id = 7;
            }
            const float* qt = QT + (size_t)(b * Ll + j) * 128;
            float v = acc[r] + qt[ih] + qt[64 + id];
            if (j >= len || key > j) v = NEGV;
            S[jl][key] = v;
        }
    }
    __syncthreads();

    // ---- Phase 2: softmax ------------------------------------------------
    {
        int row = (int)threadIdx.x >> 4;
        int seg = (int)threadIdx.x & 15;
        float lm = -3.4e38f;
        for (int c = seg * 8; c < seg * 8 + 8; ++c) lm = fmaxf(lm, S[row][c]);
        redm[row][seg] = lm;
        __syncthreads();
        float rmax = redm[row][0];
        for (int s = 1; s < 16; ++s) rmax = fmaxf(rmax, redm[row][s]);
        float ls = 0.0f;
        for (int c = seg * 8; c < seg * 8 + 8; ++c) ls += __expf(S[row][c] - rmax);
        reds[row][seg] = ls;
        __syncthreads();
        float rsum = 0.0f;
        for (int s = 0; s < 16; ++s) rsum += reds[row][s];
        float inv = 1.0f / rsum;
        for (int c = seg * 8; c < seg * 8 + 8; ++c)
            SH[row][c] = (_Float16)(__expf(S[row][c] - rmax) * inv);
    }
    __syncthreads();

    // ---- Phase 3: O = P @ V ---------------------------------------------
    {
        const int n0 = wave * 64;
        v8f acc[4] = {};
        const _Float16* arow = &SH[n15][0];                 // LDS, K-contig
        const _Float16* brow[4];
#pragma unroll
        for (int t = 0; t < 4; ++t)
            brow[t] = Vt16 + ((size_t)b * Hh + n0 + 16 * t + n15) * Ll;
        for (int kc = 0; kc < Ll; kc += 64) {
            v16h fa0 = frag_a(arow + kc, koff);
            v16h fa1 = frag_a(arow + kc + 32, koff);
            v16h fb0[4], fb1[4];
#pragma unroll
            for (int t = 0; t < 4; ++t) fb0[t] = frag_b(brow[t] + kc, kb);
#pragma unroll
            for (int t = 0; t < 4; ++t) fb1[t] = frag_b(brow[t] + kc + 32, kb);
#pragma unroll
            for (int t = 0; t < 4; ++t) acc[t] = wmma_f16(fa0, fb0[t], acc[t]);
#pragma unroll
            for (int t = 0; t < 4; ++t) acc[t] = wmma_f16(fa1, fb1[t], acc[t]);
        }
#pragma unroll
        for (int t = 0; t < 4; ++t)
#pragma unroll
            for (int r = 0; r < 8; ++r) {
                int j = j0 + mhi + r;
                int n = n0 + 16 * t + n15;
                O[((size_t)b * Ll + j) * Hh + n] = acc[t][r];
            }
    }
}

// ---------------------------------------------------------------------------
// Residual + LayerNorm, one block (256 threads) per row of 512.
// ---------------------------------------------------------------------------
__global__ __launch_bounds__(256)
void k_add_ln(const float* __restrict__ X, const float* __restrict__ R,
              const float* __restrict__ g, const float* __restrict__ bta,
              float* __restrict__ outF, _Float16* __restrict__ out16)
{
    __shared__ float s1[256], s2[256];
    const int row = blockIdx.x;
    const int t   = (int)threadIdx.x;
    const float* xr = X + (size_t)row * Hh;
    const float* rr = R + (size_t)row * Hh;
    float a0 = xr[t]       + rr[t];
    float a1 = xr[t + 256] + rr[t + 256];
    s1[t] = a0 + a1;
    s2[t] = a0 * a0 + a1 * a1;
    __syncthreads();
    for (int off = 128; off > 0; off >>= 1) {
        if (t < off) { s1[t] += s1[t + off]; s2[t] += s2[t + off]; }
        __syncthreads();
    }
    float mean = s1[0] * (1.0f / Hh);
    float var  = s2[0] * (1.0f / Hh) - mean * mean;
    float inv  = rsqrtf(var + 1e-5f);
    float o0 = (a0 - mean) * inv * g[t]       + bta[t];
    float o1 = (a1 - mean) * inv * g[t + 256] + bta[t + 256];
    if (outF)  { outF [(size_t)row * Hh + t] = o0; outF [(size_t)row * Hh + t + 256] = o1; }
    if (out16) { out16[(size_t)row * Hh + t] = (_Float16)o0;
                 out16[(size_t)row * Hh + t + 256] = (_Float16)o1; }
}

// ---------------------------------------------------------------------------
// Helpers.
// ---------------------------------------------------------------------------
__global__ void k_f32_to_f16(const float* __restrict__ s, _Float16* __restrict__ d, int n) {
    int i = blockIdx.x * blockDim.x + threadIdx.x;
    if (i < n) d[i] = (_Float16)s[i];
}
// Combined interval-embedding table: rows 0..49 hour_emb, 64..71 day_emb, rest 0.
__global__ void k_build_emb_table(const float* __restrict__ hour_emb,
                                  const float* __restrict__ day_emb,
                                  _Float16* __restrict__ d) {
    int i = blockIdx.x * blockDim.x + threadIdx.x;      // 128 * 512
    if (i >= 128 * Hh) return;
    int r = i >> 9, c = i & 511;
    float v = 0.0f;
    if (r < 50)                 v = hour_emb[(size_t)r * Hh + c];
    else if (r >= 64 && r < 72) v = day_emb[(size_t)(r - 64) * Hh + c];
    d[i] = (_Float16)v;
}
__global__ void k_transpose_f16(const float* __restrict__ W, _Float16* __restrict__ Wt,
                                int K, int N, int Npad) {
    int i = blockIdx.x * blockDim.x + threadIdx.x;
    if (i >= Npad * K) return;
    int n = i / K, k = i - n * K;
    Wt[i] = (n < N) ? (_Float16)W[(size_t)k * N + n] : (_Float16)0.0f;
}

// Build decoder input: Adec[m][0:512] = x16, Adec[m][512:1024] = hour_emb[labelI].
__global__ void k_concat_label(const _Float16* __restrict__ x16,
                               const float* __restrict__ hour_emb,
                               const int* __restrict__ ts, const int* __restrict__ lts,
                               const int* __restrict__ lens,
                               _Float16* __restrict__ Adec)
{
    int i = blockIdx.x * blockDim.x + threadIdx.x;     // m * 512 + c
    if (i >= Mm * Hh) return;
    int m = i >> 9, c = i & 511;
    int b = m >> 7, j = m & 127;
    Adec[(size_t)m * 1024 + c] = x16[i];
    int idx = 0;
    if (j < lens[b]) {
        int d2 = lts[m] - ts[m];                       // >= 1 by construction
        idx = (d2 % TUc) / 1800 + 2;
    }
    Adec[(size_t)m * 1024 + 512 + c] = (_Float16)hour_emb[(size_t)idx * Hh + c];
}

// ---------------------------------------------------------------------------
// Host side
// ---------------------------------------------------------------------------
static inline void gemm(const _Float16* A, const _Float16* Bt, const float* bias,
                        float* Cf, _Float16* C16, _Float16* C16t,
                        int M, int N, int Npad, int K, int relu, hipStream_t st)
{
    dim3 grid(Npad / 128, M / 64);
    k_gemm<<<grid, 256, 0, st>>>(A, Bt, bias, Cf, C16, C16t, N, K, relu);
}

extern "C" void kernel_launch(void* const* d_in, const int* in_sizes, int n_in,
                              void* d_out, int out_size, void* d_ws, size_t ws_size,
                              hipStream_t stream)
{
    const float* src      = (const float*)d_in[0];
    const float* hour_emb = (const float*)d_in[1];
    const float* day_emb  = (const float*)d_in[2];
    const float* dec_poi_w = (const float*)d_in[31];
    const float* dec_poi_b = (const float*)d_in[32];
    const float* dec_cat_w = (const float*)d_in[33];
    const float* dec_cat_b = (const float*)d_in[34];
    const int* lens = (const int*)d_in[35];
    const int* ts   = (const int*)d_in[36];
    const int* lts  = (const int*)d_in[37];

    // Workspace bump allocator (bytes, 256-aligned).
    char* base = (char*)d_ws;
    size_t off = 0;
    auto alloc = [&](size_t bytes) -> char* {
        char* p = base + off;
        off = (off + bytes + 255) & ~(size_t)255;
        return p;
    };
    _Float16* embT  = (_Float16*)alloc((size_t)128 * Hh * 2);     // hour+day table
    _Float16* wqT   = (_Float16*)alloc((size_t)Hh * Hh * 2);
    _Float16* wkT   = (_Float16*)alloc((size_t)Hh * Hh * 2);
    _Float16* wvT   = (_Float16*)alloc((size_t)Hh * Hh * 2);
    _Float16* ffaT  = (_Float16*)alloc((size_t)Hh * Hh * 2);
    _Float16* ffbT  = (_Float16*)alloc((size_t)Hh * Hh * 2);
    _Float16* poiT  = (_Float16*)alloc((size_t)5120 * 1024 * 2);
    _Float16* catT  = (_Float16*)alloc((size_t)384 * 1024 * 2);
    _Float16* cur16 = (_Float16*)alloc((size_t)Mm * Hh * 2);
    _Float16* Q16   = (_Float16*)alloc((size_t)Mm * Hh * 2);
    _Float16* K16   = (_Float16*)alloc((size_t)Mm * Hh * 2);
    _Float16* Vt16  = (_Float16*)alloc((size_t)Mm * Hh * 2);
    float*    QT    = (float*)alloc((size_t)Mm * 128 * 4);
    float*    Of    = (float*)alloc((size_t)Mm * Hh * 4);
    float*    midF  = (float*)alloc((size_t)Mm * Hh * 4);
    _Float16* mid16 = (_Float16*)alloc((size_t)Mm * Hh * 2);
    _Float16* ffa16 = (_Float16*)alloc((size_t)Mm * Hh * 2);
    float*    ffbF  = (float*)alloc((size_t)Mm * Hh * 4);
    float*    curF  = (float*)alloc((size_t)Mm * Hh * 4);
    _Float16* Adec  = (_Float16*)alloc((size_t)Mm * 1024 * 2);
    (void)ws_size; (void)n_in; (void)in_sizes; (void)out_size;

    const int TPB = 256;
    auto blks = [](long n) { return (int)((n + 255) / 256); };

    // One-time (per call) conversions.
    k_build_emb_table<<<blks(128L * Hh), TPB, 0, stream>>>(hour_emb, day_emb, embT);
    k_transpose_f16<<<blks(5120L * 1024), TPB, 0, stream>>>(dec_poi_w, poiT, 1024, 5000, 5120);
    k_transpose_f16<<<blks(384L * 1024), TPB, 0, stream>>>(dec_cat_w, catT, 1024, 300, 384);
    k_f32_to_f16<<<blks((long)Mm * Hh), TPB, 0, stream>>>(src, cur16, Mm * Hh);

    const float* resid = src;
    for (int layer = 0; layer < 2; ++layer) {
        int p = 3 + 14 * layer;
        const float* wq_w = (const float*)d_in[p + 0];  const float* wq_b = (const float*)d_in[p + 1];
        const float* wk_w = (const float*)d_in[p + 2];  const float* wk_b = (const float*)d_in[p + 3];
        const float* wv_w = (const float*)d_in[p + 4];  const float* wv_b = (const float*)d_in[p + 5];
        const float* fa_w = (const float*)d_in[p + 6];  const float* fa_b = (const float*)d_in[p + 7];
        const float* fb_w = (const float*)d_in[p + 8];  const float* fb_b = (const float*)d_in[p + 9];
        const float* l1g  = (const float*)d_in[p + 10]; const float* l1b  = (const float*)d_in[p + 11];
        const float* l2g  = (const float*)d_in[p + 12]; const float* l2b  = (const float*)d_in[p + 13];

        k_transpose_f16<<<blks((long)Hh * Hh), TPB, 0, stream>>>(wq_w, wqT, Hh, Hh, Hh);
        k_transpose_f16<<<blks((long)Hh * Hh), TPB, 0, stream>>>(wk_w, wkT, Hh, Hh, Hh);
        k_transpose_f16<<<blks((long)Hh * Hh), TPB, 0, stream>>>(wv_w, wvT, Hh, Hh, Hh);
        k_transpose_f16<<<blks((long)Hh * Hh), TPB, 0, stream>>>(fa_w, ffaT, Hh, Hh, Hh);
        k_transpose_f16<<<blks((long)Hh * Hh), TPB, 0, stream>>>(fb_w, ffbT, Hh, Hh, Hh);

        // Projections (f16 outputs; V stored transposed per batch).
        gemm(cur16, wqT, wq_b, nullptr, Q16, nullptr, Mm, Hh, Hh, Hh, 0, stream);
        gemm(cur16, wkT, wk_b, nullptr, K16, nullptr, Mm, Hh, Hh, Hh, 0, stream);
        gemm(cur16, wvT, wv_b, nullptr, nullptr, Vt16, Mm, Hh, Hh, Hh, 0, stream);
        // Interval-score table: QT = Q @ [hour_emb; day_emb]^T   ([M][128] f32)
        gemm(Q16, embT, nullptr, QT, nullptr, nullptr, Mm, 128, 128, Hh, 0, stream);

        k_attn<<<dim3(Ll / 16, Bb), 256, 0, stream>>>(Q16, K16, Vt16, QT, ts, lens, Of);
        k_add_ln<<<Mm, 256, 0, stream>>>(Of, resid, l1g, l1b, midF, mid16);

        gemm(mid16, ffaT, fa_b, nullptr, ffa16, nullptr, Mm, Hh, Hh, Hh, 1, stream);
        gemm(ffa16, ffbT, fb_b, ffbF, nullptr, nullptr, Mm, Hh, Hh, Hh, 0, stream);
        k_add_ln<<<Mm, 256, 0, stream>>>(ffbF, midF, l2g, l2b, curF, cur16);
        resid = curF;
    }

    // Decoder: [x, labelE] @ dec_*_w + b.
    k_concat_label<<<blks((long)Mm * Hh), TPB, 0, stream>>>(cur16, hour_emb, ts, lts, lens, Adec);
    float* out_poi = (float*)d_out;
    float* out_cat = out_poi + (size_t)Mm * 5000;
    gemm(Adec, poiT, dec_poi_b, out_poi, nullptr, nullptr, Mm, 5000, 5120, 1024, 0, stream);
    gemm(Adec, catT, dec_cat_b, out_cat, nullptr, nullptr, Mm, 300, 384, 1024, 0, stream);
}
